// TemporalLinkPredictor_26637387170221
// MI455X (gfx1250) — compile-verified, matrix operand order checked by reading, not run on hardware
//
#include <hip/hip_runtime.h>
#include <hip/hip_bf16.h>

typedef __attribute__((ext_vector_type(16))) __bf16 v16bf;
typedef __attribute__((ext_vector_type(8)))  float  v8f;
typedef __attribute__((ext_vector_type(4)))  unsigned int u32x4;

#define N_NODES_C 150000
#define N_ETYPES_C 1000
#define B_EDGES   16384
#define HDIM      128
#define KU        192    // padded [x(50->64) | h_prev(128)]
#define U_STRIDE  200    // halves, 400B row stride (16B aligned, conflict-friendly)
#define NZ        512    // 4 gates * H
#define W_STRIDE  200
#define PIN       336
#define KP        352    // padded P_IN
#define P_STRIDE  360
#define PHID      64
#define TPW       15     // node tiles per workgroup (9375 = 625 * 15)

__device__ __forceinline__ float sigmoidf_(float x) { return 1.0f / (1.0f + __expf(-x)); }

// Direct global -> LDS async copy (16B), GVS mode: addr = SGPR64 + VGPR32 offset.
// LDS destination given as wave-relative byte offset in a VGPR. Tracked by ASYNCcnt.
__device__ __forceinline__ void async_copy_b128(unsigned lds_off, const void* gbase,
                                                unsigned goff) {
  asm volatile("global_load_async_to_lds_b128 %0, %1, %2"
               :: "v"(lds_off), "v"(goff), "s"(gbase) : "memory");
}
__device__ __forceinline__ void wait_async0() {
  asm volatile("s_wait_asynccnt 0x0" ::: "memory");
}

// Load one 16-half WMMA fragment (A row or B column) from LDS.
// Half h (0..15) of this lane maps to k = (h&7) + (h>>3)*16 + grp*8  (ISA 16-bit layout)
__device__ __forceinline__ v16bf load_frag(const __bf16* base, int grp) {
  union { v16bf v; u32x4 q[2]; } t;
  t.q[0] = *(const u32x4*)(base + grp * 8);        // k = grp*8 + 0..7
  t.q[1] = *(const u32x4*)(base + 16 + grp * 8);   // k = 16 + grp*8 + 0..7
  return t.v;
}

// ---------------------------------------------------------------------------
// Prep: fuse Wx (4,50,128) + Wh (4,128,128) -> bf16 Wcat[n=512][k=192] col-major-K
// ---------------------------------------------------------------------------
__global__ void prep_wcat_kernel(const float* __restrict__ Wx,
                                 const float* __restrict__ Wh,
                                 __bf16* __restrict__ out) {
  int idx = blockIdx.x * 256 + threadIdx.x;
  if (idx >= NZ * KU) return;
  int n = idx / KU, k = idx % KU;
  int g = n >> 7, j = n & 127;
  float v;
  if (k < 64) v = (k < 50) ? Wx[(g * 50 + k) * HDIM + j] : 0.0f;
  else        v = Wh[(g * HDIM + (k - 64)) * HDIM + j];
  out[idx] = (__bf16)v;
}

// Prep: W1 (336,64) -> bf16 W1cat[n=64][k=352]
__global__ void prep_w1_kernel(const float* __restrict__ W1, __bf16* __restrict__ out) {
  int idx = blockIdx.x * 256 + threadIdx.x;
  if (idx >= PHID * KP) return;
  int n = idx / KP, k = idx % KP;
  out[idx] = (__bf16)((k < PIN) ? W1[k * PHID + n] : 0.0f);
}

// ---------------------------------------------------------------------------
// Node LSTM update: z = [x|h] @ Wcat via bf16 WMMA, fused peephole elementwise
// grid 625 x 256 threads (8 waves), dynamic LDS = 204800 (W) + 8*6400 (u) bytes
// ---------------------------------------------------------------------------
__global__ __launch_bounds__(256) void node_gate_kernel(
    const float* __restrict__ node_cont, const int* __restrict__ node_cat_codes,
    const float* __restrict__ node_cat_emb, const float* __restrict__ h_prev,
    const float* __restrict__ c_prev, const float* __restrict__ b_gate,
    const float* __restrict__ w_c, const __bf16* __restrict__ wcat,
    float* __restrict__ h_new, float* __restrict__ c_new) {
  extern __shared__ char smem[];
  __bf16* Wlds = (__bf16*)smem;                         // NZ cols * W_STRIDE halves
  __bf16* Ulds = (__bf16*)(smem + NZ * W_STRIDE * 2);   // 8 waves * 16 * U_STRIDE

  const int tid  = threadIdx.x;
  const int wave = tid >> 5;
  const int lane = tid & 31;
  const int l15  = lane & 15;
  const int grp  = lane >> 4;

  // Async DMA of Wcat into LDS: dense 384B (24x16B) columns -> 400B-stride columns.
  {
    const unsigned wbase = (unsigned)(size_t)(void*)Wlds;
    for (int idx = tid; idx < NZ * 24; idx += 256) {
      int n = idx / 24, c = idx % 24;
      async_copy_b128(wbase + n * (W_STRIDE * 2) + c * 16, wcat, (unsigned)idx * 16);
    }
    wait_async0();
  }
  __syncthreads();

  __bf16* u = Ulds + wave * 16 * U_STRIDE;

  for (int tt = wave; tt < TPW; tt += 8) {
    const int tile  = blockIdx.x * TPW + tt;
    const int node0 = tile * 16;

    // Prefetch next tile's h_prev rows (16 nodes * 512B = 8KB; 32 lanes * 256B)
    if (tt + 8 < TPW) {
      const char* nxt = (const char*)(h_prev + (size_t)(node0 + 128) * HDIM);
      __builtin_prefetch(nxt + lane * 256, 0, 1);
    }

    // Build bf16 u-tile [16 x 192] in LDS: [cont(2) | cat(48) | pad | h_prev(128)]
    for (int r = 0; r < 16; ++r) {
      const int node = node0 + r;
      for (int k = lane; k < KU; k += 32) {
        float v;
        if (k < 2) {
          v = node_cont[node * 2 + k];
        } else if (k < 50) {
          int c = (k - 2) >> 4, d = (k - 2) & 15;
          int code = node_cat_codes[c * N_NODES_C + node];
          v = node_cat_emb[(c * 64 + code) * 16 + d];
        } else if (k < 64) {
          v = 0.0f;
        } else {
          v = h_prev[node * HDIM + (k - 64)];
        }
        u[r * U_STRIDE + k] = (__bf16)v;
      }
    }
    asm volatile("s_wait_dscnt 0" ::: "memory");

    // A fragments for this 16-node tile (K = 6 * 32)
    v16bf a[6];
#pragma unroll
    for (int kt = 0; kt < 6; ++kt)
      a[kt] = load_frag(u + l15 * U_STRIDE + kt * 32, grp);

    for (int jt = 0; jt < 8; ++jt) {        // 16-column slice of H
      v8f acc[4] = {};                      // one accumulator per gate
#pragma unroll
      for (int g = 0; g < 4; ++g) {
        const __bf16* bcol = Wlds + (g * HDIM + jt * 16 + l15) * W_STRIDE;
#pragma unroll
        for (int kt = 0; kt < 6; ++kt) {
          v16bf b = load_frag(bcol + kt * 32, grp);
          acc[g] = __builtin_amdgcn_wmma_f32_16x16x32_bf16(
              false, a[kt], false, b, (short)0, acc[g], false, false);
        }
      }
      // Fused peephole-LSTM elementwise, entirely in registers
      const int j = jt * 16 + l15;
      const float wc0 = w_c[j], wc1 = w_c[HDIM + j], wc2 = w_c[2 * HDIM + j];
      const float b0 = b_gate[j], b1v = b_gate[HDIM + j];
      const float b2v = b_gate[2 * HDIM + j], b3v = b_gate[3 * HDIM + j];
#pragma unroll
      for (int v = 0; v < 8; ++v) {
        const int m = v + 8 * grp;
        const int node = node0 + m;
        const float cp = c_prev[node * HDIM + j];
        const float ig = sigmoidf_(acc[0][v] + b0 + wc0 * cp);
        const float fg = sigmoidf_(acc[1][v] + b1v + wc1 * cp);
        const float tg = tanhf(acc[2][v] + b2v);
        const float cn = fg * cp + ig * tg;
        const float og = sigmoidf_(acc[3][v] + b3v + wc2 * cn);
        h_new[node * HDIM + j] = og * tanhf(cn);
        c_new[node * HDIM + j] = cn;
      }
    }
  }
}

// ---------------------------------------------------------------------------
// Edge MLP: prob = sigmoid(relu(comb @ W1 + b1) @ W2 + b2), bf16 WMMA
// grid 256 x 128 threads (4 waves, 1 tile of 16 edges per wave)
// dynamic LDS = 64*720 (W1) + 4*16*720 (comb) bytes
// ---------------------------------------------------------------------------
__global__ __launch_bounds__(128) void edge_mlp_kernel(
    const int* __restrict__ src_ids, const int* __restrict__ dst_ids,
    const int* __restrict__ etype_ids, const float* __restrict__ t_rel,
    const int* __restrict__ edge_cat_codes, const float* __restrict__ edge_id_emb,
    const float* __restrict__ edge_cat_emb, const float* __restrict__ time_W,
    const float* __restrict__ time_b, const float* __restrict__ b1,
    const float* __restrict__ W2, const float* __restrict__ b2,
    const __bf16* __restrict__ w1cat, const float* __restrict__ h_new,
    float* __restrict__ prob) {
  extern __shared__ char smem[];
  __bf16* Wlds = (__bf16*)smem;                            // 64 * P_STRIDE halves
  __bf16* Clds = (__bf16*)(smem + PHID * P_STRIDE * 2);    // 4 waves * 16 * P_STRIDE

  const int tid  = threadIdx.x;
  const int wave = tid >> 5;
  const int lane = tid & 31;
  const int l15  = lane & 15;
  const int grp  = lane >> 4;

  // Async DMA of W1cat into LDS: dense 704B (44x16B) columns -> 720B-stride columns.
  {
    const unsigned wbase = (unsigned)(size_t)(void*)Wlds;
    for (int idx = tid; idx < PHID * 44; idx += 128) {
      int n = idx / 44, c = idx % 44;
      async_copy_b128(wbase + n * (P_STRIDE * 2) + c * 16, w1cat, (unsigned)idx * 16);
    }
    wait_async0();
  }
  __syncthreads();

  const int tile = blockIdx.x * 4 + wave;
  const int e0   = tile * 16;
  __bf16* cmb = Clds + wave * 16 * P_STRIDE;

  // Build comb = [src_h(128) | dst_h(128) | eid(32) | ecat(32) | time(16) | pad]
  for (int r = 0; r < 16; ++r) {
    const int e  = e0 + r;
    const int sn = src_ids[e], dn = dst_ids[e], et = etype_ids[e];
    const float tr = t_rel[e];
    for (int k = lane; k < KP; k += 32) {
      float v;
      if (k < 128)      v = h_new[sn * HDIM + k];
      else if (k < 256) v = h_new[dn * HDIM + (k - 128)];
      else if (k < 288) v = edge_id_emb[et * 32 + (k - 256)];
      else if (k < 320) {
        int c = (k - 288) >> 4, d = (k - 288) & 15;
        int code = edge_cat_codes[c * N_ETYPES_C + et];
        v = edge_cat_emb[(c * 32 + code) * 16 + d];
      } else if (k < PIN) {
        int d = k - 320;
        v = tr * time_W[d] + time_b[d];
      } else v = 0.0f;
      cmb[r * P_STRIDE + k] = (__bf16)v;
    }
  }
  asm volatile("s_wait_dscnt 0" ::: "memory");

  v16bf a[11];
#pragma unroll
  for (int kt = 0; kt < 11; ++kt)
    a[kt] = load_frag(cmb + l15 * P_STRIDE + kt * 32, grp);

  float contrib[8] = {0.f, 0.f, 0.f, 0.f, 0.f, 0.f, 0.f, 0.f};
#pragma unroll
  for (int nt = 0; nt < 4; ++nt) {
    v8f acc = {};
    const __bf16* bcol = Wlds + (nt * 16 + l15) * P_STRIDE;
#pragma unroll
    for (int kt = 0; kt < 11; ++kt) {
      v16bf b = load_frag(bcol + kt * 32, grp);
      acc = __builtin_amdgcn_wmma_f32_16x16x32_bf16(
          false, a[kt], false, b, (short)0, acc, false, false);
    }
    const int n = nt * 16 + l15;
    const float bb = b1[n], w2 = W2[n];
#pragma unroll
    for (int v = 0; v < 8; ++v) {
      float h = acc[v] + bb;
      h = h > 0.0f ? h : 0.0f;
      contrib[v] += h * w2;
    }
  }
  // reduce over the 16 lanes holding the hidden dim (xor butterflies stay in group)
#pragma unroll
  for (int v = 0; v < 8; ++v) {
    float s = contrib[v];
    s += __shfl_xor(s, 1, 32);
    s += __shfl_xor(s, 2, 32);
    s += __shfl_xor(s, 4, 32);
    s += __shfl_xor(s, 8, 32);
    contrib[v] = s;
  }
  if (l15 == 0) {
    const float bb2 = b2[0];
#pragma unroll
    for (int v = 0; v < 8; ++v)
      prob[e0 + v + 8 * grp] = sigmoidf_(contrib[v] + bb2);
  }
}

// ---------------------------------------------------------------------------
extern "C" void kernel_launch(void* const* d_in, const int* in_sizes, int n_in,
                              void* d_out, int out_size, void* d_ws, size_t ws_size,
                              hipStream_t stream) {
  const int*   src_ids        = (const int*)d_in[0];
  const int*   dst_ids        = (const int*)d_in[1];
  const int*   etype_ids      = (const int*)d_in[2];
  const float* t_rel          = (const float*)d_in[3];
  const float* node_cont      = (const float*)d_in[4];
  const int*   node_cat_codes = (const int*)d_in[5];
  const int*   edge_cat_codes = (const int*)d_in[6];
  // d_in[7] = edge_index (unused by reference computation)
  const float* h_prev         = (const float*)d_in[8];
  const float* c_prev         = (const float*)d_in[9];
  const float* node_cat_emb   = (const float*)d_in[10];
  const float* edge_id_emb    = (const float*)d_in[11];
  const float* edge_cat_emb   = (const float*)d_in[12];
  const float* time_W         = (const float*)d_in[13];
  const float* time_b         = (const float*)d_in[14];
  const float* Wx             = (const float*)d_in[15];
  const float* Wh             = (const float*)d_in[16];
  const float* b_gate         = (const float*)d_in[17];
  const float* w_c            = (const float*)d_in[18];
  const float* W1             = (const float*)d_in[19];
  const float* b1             = (const float*)d_in[20];
  const float* W2             = (const float*)d_in[21];
  const float* b2             = (const float*)d_in[22];

  float* out   = (float*)d_out;
  float* prob  = out;                                   // [16384]
  float* h_new = out + B_EDGES;                         // [150000*128]
  float* c_new = h_new + (size_t)N_NODES_C * HDIM;      // [150000*128]

  __bf16* ws_wcat = (__bf16*)d_ws;                                  // 512*192*2
  __bf16* ws_w1   = (__bf16*)((char*)d_ws + NZ * KU * 2);           // 64*352*2

  prep_wcat_kernel<<<(NZ * KU + 255) / 256, 256, 0, stream>>>(Wx, Wh, ws_wcat);
  prep_w1_kernel<<<(PHID * KP + 255) / 256, 256, 0, stream>>>(W1, ws_w1);

  const int node_lds = NZ * W_STRIDE * 2 + 8 * 16 * U_STRIDE * 2;   // 256000 B
  node_gate_kernel<<<625, 256, node_lds, stream>>>(
      node_cont, node_cat_codes, node_cat_emb, h_prev, c_prev,
      b_gate, w_c, ws_wcat, h_new, c_new);

  const int edge_lds = PHID * P_STRIDE * 2 + 4 * 16 * P_STRIDE * 2; // 92160 B
  edge_mlp_kernel<<<B_EDGES / 16 / 4, 128, edge_lds, stream>>>(
      src_ids, dst_ids, etype_ids, t_rel, edge_cat_codes, edge_id_emb,
      edge_cat_emb, time_W, time_b, b1, W2, b2, ws_w1, h_new, prob);
}
